// Memory_cell_6957847019562
// MI455X (gfx1250) — matrix operand order repaired and checked by reading, not compiled, not run on hardware
//
#include <hip/hip_runtime.h>
#include <hip/hip_bf16.h>

typedef __attribute__((ext_vector_type(16))) __bf16 v16bf;
typedef __attribute__((ext_vector_type(8)))  __bf16 v8bf;
typedef __attribute__((ext_vector_type(8)))  float  v8f;

// Problem sizes (fixed by the reference).
constexpr int Bn = 2048;   // batch
constexpr int K  = 4096;   // memory slots
constexpr int R  = 2048;   // hidden
constexpr int V  = 2048;   // att width

// workspace layout (bytes)
constexpr size_t WS_SCORES = 0;                         // K floats
constexpr size_t WS_ATT    = WS_SCORES + K * 4;         // K floats
constexpr size_t WS_RVEC   = WS_ATT + K * 4;            // R floats
constexpr size_t WS_MH     = 64 * 1024;                 // K*R bf16 (16 MB)
constexpr size_t WS_WMH    = WS_MH + (size_t)K * R * 2; // V*R bf16 (8 MB)
constexpr size_t WS_NEED   = WS_WMH + (size_t)V * R * 2;

// ---------------------------------------------------------------------------
// helpers
// ---------------------------------------------------------------------------
__device__ __forceinline__ float fast_tanh(float x) {
    // tanh(x) = 1 - 2/(exp(2x)+1);  exp(2x) = exp2(2*log2(e)*x)
    float e = __builtin_amdgcn_exp2f(x * 2.885390081777927f);
    return 1.0f - 2.0f * __builtin_amdgcn_rcpf(e + 1.0f);
}

// fused epilogue + reduction + atomic scatter shared by both GEMM kernels
__device__ __forceinline__ void score_epilogue(const v8f* acc, int lane, int lm,
                                               int hi, int k0, int v0,
                                               const float* __restrict__ WMb,
                                               const float* __restrict__ Ww,
                                               float* __restrict__ scores) {
    float srow[8];
#pragma unroll
    for (int i = 0; i < 8; ++i) srow[i] = 0.0f;

#pragma unroll
    for (int ct = 0; ct < 16; ++ct) {
        const int n = v0 + ct * 16 + lm;
        const float wv = Ww[n];
        const float bb = WMb[n];
#pragma unroll
        for (int i = 0; i < 8; ++i)
            srow[i] += fast_tanh(acc[ct][i] + bb) * wv;
    }

    // reduce across the 16 lanes of each half (rows i / i+8)
#pragma unroll
    for (int m = 1; m < 16; m <<= 1)
#pragma unroll
        for (int i = 0; i < 8; ++i)
            srow[i] += __shfl_xor(srow[i], m, 32);

    if (lm == 0) {
        const int rowbase = k0 + hi * 8;   // lane0 -> rows 0..7, lane16 -> rows 8..15
#pragma unroll
        for (int i = 0; i < 8; ++i)
            atomicAdd(&scores[rowbase + i], srow[i]);
    }
}

// ---------------------------------------------------------------------------
// K0: zero scratch (scores[K] and r[R])
// ---------------------------------------------------------------------------
__global__ void init_ws_kernel(float* scores, float* rvec) {
    int i = blockIdx.x * blockDim.x + threadIdx.x;
    if (i < K) scores[i] = 0.0f;
    if (i < R) rvec[i] = 0.0f;
}

// ---------------------------------------------------------------------------
// K0b: fp32 -> bf16 pre-conversion of M (K*R) then WM_w (V*R), 8 elems/thread
// ---------------------------------------------------------------------------
__global__ __launch_bounds__(256)
void cvt_bf16_kernel(const float* __restrict__ M, const float* __restrict__ WMw,
                     __bf16* __restrict__ Mh, __bf16* __restrict__ WMh) {
    const long i = (long)blockIdx.x * 256 + threadIdx.x;   // per 8 elements
    const long nM = (long)K * R / 8;
    const float4* src;
    uint4* dst;
    if (i < nM) {
        src = (const float4*)M + 2 * i;
        dst = (uint4*)Mh + i;
    } else {
        const long j = i - nM;
        src = (const float4*)WMw + 2 * j;
        dst = (uint4*)WMh + j;
    }
    float4 x = src[0], y = src[1];
    union { __bf16 b[8]; uint4 u; } o;
    o.b[0] = (__bf16)x.x; o.b[1] = (__bf16)x.y; o.b[2] = (__bf16)x.z; o.b[3] = (__bf16)x.w;
    o.b[4] = (__bf16)y.x; o.b[5] = (__bf16)y.y; o.b[6] = (__bf16)y.z; o.b[7] = (__bf16)y.w;
    *dst = o.u;
}

// ---------------------------------------------------------------------------
// K1 (fast path): scores from pre-converted bf16 operands.
// One wave computes a 16-row x 256-col strip of tanh(M@WM_w^T+b)·W_w.
// grid = (K/16, V/256), block = 32
// ---------------------------------------------------------------------------
__global__ __launch_bounds__(32)
void scores_bf16_kernel(const __bf16* __restrict__ Mh,
                        const __bf16* __restrict__ WMh,
                        const float* __restrict__ WMb,
                        const float* __restrict__ Ww,
                        float* __restrict__ scores) {
    const int lane = threadIdx.x;
    const int lm   = lane & 15;
    const int hi   = lane >> 4;
    const int k0   = blockIdx.x * 16;
    const int v0   = blockIdx.y * 256;

    v8f acc[16];
#pragma unroll
    for (int ct = 0; ct < 16; ++ct)
#pragma unroll
        for (int i = 0; i < 8; ++i) acc[ct][i] = 0.0f;

    const int kbaseA = hi * 16;   // A: low lanes K=0..15, high lanes K=16..31
    const int koffB  = hi * 16;   // B: low lanes K=0..15, high lanes K=16..31

    const __bf16* arowbase = Mh + (k0 + lm) * R;

    for (int r0 = 0; r0 < R; r0 += 32) {
        // ---- A fragment: two 16-byte bf16 loads, no conversion ----
        v8bf alo = *(const v8bf*)(arowbase + r0 + kbaseA);
        v8bf ahi = *(const v8bf*)(arowbase + r0 + 8 + kbaseA);
        v16bf av;
#pragma unroll
        for (int j = 0; j < 8; ++j) { av[j] = alo[j]; av[8 + j] = ahi[j]; }

#pragma unroll
        for (int ct = 0; ct < 16; ++ct) {
            const int n = v0 + ct * 16 + lm;
            // ---- B fragment: one contiguous 32-byte bf16 load per lane ----
            v16bf bv = *(const v16bf*)(WMh + (size_t)n * R + r0 + koffB);
            acc[ct] = __builtin_amdgcn_wmma_f32_16x16x32_bf16(
                false, av, false, bv, (short)0, acc[ct], false, false);
        }
    }

    score_epilogue(acc, lane, lm, hi, k0, v0, WMb, Ww, scores);
}

// ---------------------------------------------------------------------------
// K1 (fallback path, small ws): fused fp32->bf16 conversion inside the loop
// ---------------------------------------------------------------------------
__global__ __launch_bounds__(32)
void scores_fused_kernel(const float* __restrict__ M,
                         const float* __restrict__ WMw,
                         const float* __restrict__ WMb,
                         const float* __restrict__ Ww,
                         float* __restrict__ scores) {
    const int lane = threadIdx.x;
    const int lm   = lane & 15;
    const int hi   = lane >> 4;
    const int k0   = blockIdx.x * 16;
    const int v0   = blockIdx.y * 256;

    v8f acc[16];
#pragma unroll
    for (int ct = 0; ct < 16; ++ct)
#pragma unroll
        for (int i = 0; i < 8; ++i) acc[ct][i] = 0.0f;

    const int kbaseA = hi * 16;
    const int koffB  = hi * 16;

    for (int r0 = 0; r0 < R; r0 += 32) {
        const float* arow = M + (k0 + lm) * R + r0 + kbaseA;
        float4 a0 = *(const float4*)(arow + 0);
        float4 a1 = *(const float4*)(arow + 4);
        float4 a2 = *(const float4*)(arow + 8);
        float4 a3 = *(const float4*)(arow + 12);
        v16bf av;
        av[0]=(__bf16)a0.x; av[1]=(__bf16)a0.y; av[2]=(__bf16)a0.z; av[3]=(__bf16)a0.w;
        av[4]=(__bf16)a1.x; av[5]=(__bf16)a1.y; av[6]=(__bf16)a1.z; av[7]=(__bf16)a1.w;
        av[8]=(__bf16)a2.x; av[9]=(__bf16)a2.y; av[10]=(__bf16)a2.z; av[11]=(__bf16)a2.w;
        av[12]=(__bf16)a3.x; av[13]=(__bf16)a3.y; av[14]=(__bf16)a3.z; av[15]=(__bf16)a3.w;

#pragma unroll
        for (int ct = 0; ct < 16; ++ct) {
            const int n = v0 + ct * 16 + lm;
            const float* brow = WMw + (size_t)n * R + r0 + koffB;
            float4 b0 = *(const float4*)(brow + 0);
            float4 b1 = *(const float4*)(brow + 4);
            float4 b2 = *(const float4*)(brow + 8);
            float4 b3 = *(const float4*)(brow + 12);
            v16bf bv;
            bv[0]=(__bf16)b0.x; bv[1]=(__bf16)b0.y; bv[2]=(__bf16)b0.z; bv[3]=(__bf16)b0.w;
            bv[4]=(__bf16)b1.x; bv[5]=(__bf16)b1.y; bv[6]=(__bf16)b1.z; bv[7]=(__bf16)b1.w;
            bv[8]=(__bf16)b2.x; bv[9]=(__bf16)b2.y; bv[10]=(__bf16)b2.z; bv[11]=(__bf16)b2.w;
            bv[12]=(__bf16)b3.x; bv[13]=(__bf16)b3.y; bv[14]=(__bf16)b3.z; bv[15]=(__bf16)b3.w;

            acc[ct] = __builtin_amdgcn_wmma_f32_16x16x32_bf16(
                false, av, false, bv, (short)0, acc[ct], false, false);
        }
    }

    score_epilogue(acc, lane, lm, hi, k0, v0, WMb, Ww, scores);
}

// ---------------------------------------------------------------------------
// K2: att = softmax(scores)  (single block, K=4096, 1024 threads, 4/thread)
// ---------------------------------------------------------------------------
__global__ __launch_bounds__(1024)
void softmax_kernel(const float* __restrict__ scores, float* __restrict__ att) {
    __shared__ float red[32];
    const int t = threadIdx.x;
    const int wid = t >> 5, lane = t & 31;

    float v[4];
    float mx = -1e30f;
#pragma unroll
    for (int i = 0; i < 4; ++i) {
        v[i] = scores[t + i * 1024];
        mx = fmaxf(mx, v[i]);
    }
#pragma unroll
    for (int m = 1; m < 32; m <<= 1) mx = fmaxf(mx, __shfl_xor(mx, m, 32));
    if (lane == 0) red[wid] = mx;
    __syncthreads();
    if (wid == 0) {
        float x = red[lane];
#pragma unroll
        for (int m = 1; m < 32; m <<= 1) x = fmaxf(x, __shfl_xor(x, m, 32));
        red[lane] = x;
    }
    __syncthreads();
    mx = red[0];

    float s = 0.0f;
#pragma unroll
    for (int i = 0; i < 4; ++i) {
        v[i] = __builtin_amdgcn_exp2f((v[i] - mx) * 1.4426950408889634f);
        s += v[i];
    }
#pragma unroll
    for (int m = 1; m < 32; m <<= 1) s += __shfl_xor(s, m, 32);
    __syncthreads();
    if (lane == 0) red[wid] = s;
    __syncthreads();
    if (wid == 0) {
        float x = red[lane];
#pragma unroll
        for (int m = 1; m < 32; m <<= 1) x += __shfl_xor(x, m, 32);
        red[lane] = x;
    }
    __syncthreads();
    const float inv = 1.0f / red[0];
#pragma unroll
    for (int i = 0; i < 4; ++i) att[t + i * 1024] = v[i] * inv;
}

// ---------------------------------------------------------------------------
// K3: r[rcol] += sum_{k in chunk} att[k] * M[k, rcol]
// grid = (R/256, K/1024), block = 256; coalesced column reads of M
// ---------------------------------------------------------------------------
__global__ __launch_bounds__(256)
void attres_kernel(const float* __restrict__ M,
                   const float* __restrict__ att,
                   float* __restrict__ rvec) {
    const int rcol = blockIdx.x * 256 + threadIdx.x;
    const int kb   = blockIdx.y * 1024;
    float acc = 0.0f;
    for (int k = kb; k < kb + 1024; ++k)
        acc += att[k] * M[(size_t)k * R + rcol];
    atomicAdd(&rvec[rcol], acc);
}

// ---------------------------------------------------------------------------
// K4: out[b, :] = r  for all b   (float4 broadcast, B*R/4 elements)
// ---------------------------------------------------------------------------
__global__ __launch_bounds__(256)
void broadcast_kernel(const float* __restrict__ rvec, float* __restrict__ out) {
    const int idx = blockIdx.x * 256 + threadIdx.x;   // over Bn*R/4
    const int r4  = idx & (R / 4 - 1);
    const float4 val = ((const float4*)rvec)[r4];
    ((float4*)out)[idx] = val;
}

// ---------------------------------------------------------------------------
// launcher
// ---------------------------------------------------------------------------
extern "C" void kernel_launch(void* const* d_in, const int* in_sizes, int n_in,
                              void* d_out, int out_size, void* d_ws, size_t ws_size,
                              hipStream_t stream) {
    // setup_inputs order: h, M, Wh_w, Wh_b, WM_w, WM_b, W_w, W_b
    const float* M   = (const float*)d_in[1];
    const float* WMw = (const float*)d_in[4];
    const float* WMb = (const float*)d_in[5];
    const float* Ww  = (const float*)d_in[6];
    float* out = (float*)d_out;

    char* wsb = (char*)d_ws;
    float* scores = (float*)(wsb + WS_SCORES);
    float* att    = (float*)(wsb + WS_ATT);
    float* rvec   = (float*)(wsb + WS_RVEC);

    init_ws_kernel<<<(K + 255) / 256, 256, 0, stream>>>(scores, rvec);

    if (ws_size >= WS_NEED) {
        // fast path: one-shot bf16 conversion, then pure-bf16 WMMA GEMM
        __bf16* Mh  = (__bf16*)(wsb + WS_MH);
        __bf16* WMh = (__bf16*)(wsb + WS_WMH);
        const long n8 = ((long)K * R + (long)V * R) / 8;
        cvt_bf16_kernel<<<(int)(n8 / 256), 256, 0, stream>>>(M, WMw, Mh, WMh);
        scores_bf16_kernel<<<dim3(K / 16, V / 256), 32, 0, stream>>>(
            Mh, WMh, WMb, Ww, scores);
    } else {
        // fallback: convert inside the GEMM loop
        scores_fused_kernel<<<dim3(K / 16, V / 256), 32, 0, stream>>>(
            M, WMw, WMb, Ww, scores);
    }

    softmax_kernel<<<1, 1024, 0, stream>>>(scores, att);

    attres_kernel<<<dim3(R / 256, K / 1024), 256, 0, stream>>>(M, att, rvec);

    broadcast_kernel<<<(Bn * R / 4) / 256, 256, 0, stream>>>(rvec, out);
}
